// WindowAttention_31860067402154
// MI455X (gfx1250) — compile-verified, matrix operand order checked by reading, not run on hardware
//
#include <hip/hip_runtime.h>
#include <math.h>

// ---- problem constants ----
#define BATCH 2048
#define NTOK  64
#define CDIM  256
#define HEADS 8
#define HD    32

typedef __attribute__((ext_vector_type(16))) _Float16 v16h;
typedef __attribute__((ext_vector_type(8)))  _Float16 v8h;
typedef __attribute__((ext_vector_type(8)))  float    v8f;

// Load one WMMA 16x16x32 f16 operand fragment for this lane from row-major
// storage (rowPtr = start of this lane's row; stride handled by caller).
// ISA layout: elems 0..7 -> K = hlf*8 + i ; elems 8..15 -> K = 16 + hlf*8 + i.
__device__ __forceinline__ v16h load_frag(const _Float16* rowPtr, int kBase, int hlf) {
    v8h lo = *(const v8h*)(rowPtr + kBase + hlf * 8);
    v8h hi = *(const v8h*)(rowPtr + kBase + 16 + hlf * 8);
    v16h r;
#pragma unroll
    for (int i = 0; i < 8; ++i) { r[i] = lo[i]; r[8 + i] = hi[i]; }
    return r;
}

__device__ __forceinline__ v8f wmma_f16(v16h a, v16h b, v8f c) {
    return __builtin_amdgcn_wmma_f32_16x16x32_f16(false, a, false, b, (short)0, c, false, false);
}

// ---------------------------------------------------------------------------
// Kernel 1: continuous position bias table.  biasTab[h][n][m] = 16*sigmoid(
//   MLP(rel_coords_table)[rel_pos_index[n][m]][h] ),  MLP = relu(x@w1^T+b1)@w2^T
// ---------------------------------------------------------------------------
__global__ void cpb_kernel(const float* __restrict__ tbl2,   // [225][2]
                           const float* __restrict__ w1,     // [512][2]
                           const float* __restrict__ b1,     // [512]
                           const float* __restrict__ w2,     // [8][512]
                           const int*   __restrict__ idx,    // [64][64]
                           float* __restrict__ biasTab) {    // [8][64][64]
    __shared__ float t[225 * 8];
    const int tid = threadIdx.x;
    for (int p = tid; p < 225 * 8; p += 256) {
        int e = p >> 3, h = p & 7;
        float i0 = tbl2[e * 2 + 0], i1 = tbl2[e * 2 + 1];
        float acc = 0.0f;
        for (int j = 0; j < 512; ++j) {
            float hv = fmaxf(w1[j * 2 + 0] * i0 + w1[j * 2 + 1] * i1 + b1[j], 0.0f);
            acc += w2[h * 512 + j] * hv;
        }
        t[p] = acc;
    }
    __syncthreads();
    for (int e = tid; e < 8 * 4096; e += 256) {
        int h = e >> 12, ij = e & 4095;
        float val = t[idx[ij] * 8 + h];
        biasTab[e] = 16.0f / (1.0f + __expf(-val));
    }
}

// ---------------------------------------------------------------------------
// Kernel 2: QKV GEMM + bias + per-head L2 normalization (sections: 0=q,1=k,2=v)
// grid = (2048, 3), block = 256 (8 waves).  Block tile: 64 rows x 256 cols.
// Per-wave tile: 32 rows x 64 cols (2 A-frags x 4 B-frags -> 8 WMMAs; each
// B fragment feeds 2 WMMAs to amortize ds_load waits).
// Outputs f16 in [B][H][N][HD] layout.
// ---------------------------------------------------------------------------
__global__ void qkv_kernel(const float* __restrict__ x,        // [B*N][256]
                           const float* __restrict__ qkv_w,    // [768][256]
                           const float* __restrict__ q_bias,   // [256]
                           const float* __restrict__ v_bias,   // [256]
                           _Float16* __restrict__ qbuf,
                           _Float16* __restrict__ kbuf,
                           _Float16* __restrict__ vbuf) {
    extern __shared__ char smem[];
    _Float16* xs  = (_Float16*)smem;            // [64][40] f16
    _Float16* wsm = xs + 64 * 40;               // [256][40] f16
    const int sec  = blockIdx.y;
    const int tid  = threadIdx.x;
    const int lane = tid & 31, wave = tid >> 5;
    const int m16 = lane & 15, hlf = lane >> 4;
    const int rp = wave & 1;           // row pair  (32 rows)
    const int cq = wave >> 1;          // col quarter (64 cols)
    const long rowBase = (long)blockIdx.x * 64;

    v8f acc[8];
#pragma unroll
    for (int t = 0; t < 8; ++t) acc[t] = (v8f){0, 0, 0, 0, 0, 0, 0, 0};

    for (int k0 = 0; k0 < 256; k0 += 32) {
        // stage x tile (64x32 f32 -> f16), prefetch next K-chunk
        for (int c = tid; c < 512; c += 256) {
            int r = c >> 3, cc = (c & 7) * 4;
            const float* src = x + (rowBase + r) * 256 + k0 + cc;
            float4 f = *(const float4*)src;
            if (k0 + 32 < 256) __builtin_prefetch(src + 32, 0, 3);
            _Float16* d = xs + r * 40 + cc;
            d[0] = (_Float16)f.x; d[1] = (_Float16)f.y;
            d[2] = (_Float16)f.z; d[3] = (_Float16)f.w;
        }
        // stage weight tile (256 output cols x 32 K, f32 -> f16)
        for (int c = tid; c < 2048; c += 256) {
            int r = c >> 3, cc = (c & 7) * 4;
            const float* src = qkv_w + (long)(sec * 256 + r) * 256 + k0 + cc;
            float4 f = *(const float4*)src;
            if (k0 + 32 < 256) __builtin_prefetch(src + 32, 0, 3);
            _Float16* d = wsm + r * 40 + cc;
            d[0] = (_Float16)f.x; d[1] = (_Float16)f.y;
            d[2] = (_Float16)f.z; d[3] = (_Float16)f.w;
        }
        __syncthreads();
        v16h a0 = load_frag(xs + (rp * 32 + m16) * 40, 0, hlf);
        v16h a1 = load_frag(xs + (rp * 32 + 16 + m16) * 40, 0, hlf);
#pragma unroll
        for (int t = 0; t < 4; ++t) {
            v16h b = load_frag(wsm + (cq * 64 + t * 16 + m16) * 40, 0, hlf);
            acc[t]     = wmma_f16(a0, b, acc[t]);
            acc[4 + t] = wmma_f16(a1, b, acc[4 + t]);
        }
        __syncthreads();
    }

    // Epilogue: bias add, per-(row, head) L2 norm for q/k, f16 writeout.
    // obuf overlays staging LDS; processed in two col-half passes (<64KB LDS).
    float* obuf = (float*)smem;                 // [64][132] f32
    _Float16* dstbuf = (sec == 0) ? qbuf : (sec == 1) ? kbuf : vbuf;
    for (int pass = 0; pass < 2; ++pass) {
        if ((cq >> 1) == pass) {
#pragma unroll
            for (int j = 0; j < 8; ++j) {
                int cl = (cq & 1) * 64 + (j & 3) * 16 + m16;
                int colG = pass * 128 + cl;
                float bv = (sec == 0) ? q_bias[colG] : (sec == 2 ? v_bias[colG] : 0.0f);
#pragma unroll
                for (int v = 0; v < 8; ++v) {
                    int r = rp * 32 + (j >> 2) * 16 + v + 8 * hlf;
                    obuf[r * 132 + cl] = acc[j][v] + bv;
                }
            }
        }
        __syncthreads();
        {
            int r = tid >> 2, hl = tid & 3;        // 64 rows x 4 heads
            int hh = pass * 4 + hl;
            const float* rowp = obuf + r * 132 + hl * 32;
            float rs = 1.0f;
            if (sec < 2) {
                float ss = 0.0f;
                for (int dd = 0; dd < 32; ++dd) { float vv = rowp[dd]; ss += vv * vv; }
                rs = 1.0f / fmaxf(sqrtf(ss), 1e-12f);
            }
            _Float16* dp = dstbuf + (((long)blockIdx.x * 8 + hh) * 64 + r) * 32;
            for (int dd = 0; dd < 32; ++dd) dp[dd] = (_Float16)(rowp[dd] * rs);
        }
        __syncthreads();
    }
}

// ---------------------------------------------------------------------------
// Kernel 3: attention per (window, head).  grid = (2048, 8), block = 128.
// S = q@k^T (K=32, one WMMA per tile) * exp(min(scale,ln100)) + cpb bias,
// fp32 softmax, out = attn(f16) @ v via 2 WMMA K-steps.  Writes [B][N][256] f16.
// ---------------------------------------------------------------------------
__global__ void attn_kernel(const _Float16* __restrict__ qbuf,
                            const _Float16* __restrict__ kbuf,
                            const _Float16* __restrict__ vbuf,
                            const float* __restrict__ biasTab,   // [8][64][64]
                            const float* __restrict__ scale,     // [8]
                            _Float16* __restrict__ attnout) {    // [B*N][256]
    extern __shared__ char smem[];
    _Float16* qs  = (_Float16*)smem;            // [64][40]
    _Float16* ks  = qs + 64 * 40;               // [64][40]
    _Float16* vsT = ks + 64 * 40;               // [32][72]  (v transposed: [d][m])
    float*    sb  = (float*)(smem + 14848);     // [64][66] f32
    _Float16* af  = (_Float16*)(smem + 31744);  // [64][72] f16 softmax probs

    const int b = blockIdx.x, h = blockIdx.y;
    const int tid = threadIdx.x, lane = tid & 31, wave = tid >> 5;
    const int m16 = lane & 15, hlf = lane >> 4;
    const long base = ((long)b * 8 + h) * 64 * 32;

    // stage q, k (vectorized) and v transposed (scalar scatter)
    for (int c = tid; c < 256; c += 128) {
        int r = c >> 2, cc = (c & 3) * 8;
        *(v8h*)(qs + r * 40 + cc) = *(const v8h*)(qbuf + base + r * 32 + cc);
        *(v8h*)(ks + r * 40 + cc) = *(const v8h*)(kbuf + base + r * 32 + cc);
    }
    for (int e = tid; e < 2048; e += 128) {
        int r = e >> 5, dd = e & 31;
        vsT[dd * 72 + r] = vbuf[base + r * 32 + dd];
    }
    __syncthreads();

    const float sc = __expf(fminf(scale[h], 4.60517018599f));   // ln(100)
    const float* bt = biasTab + h * 4096;

    // S = q @ k^T  (each wave owns one 16-row n-block, all 4 m-blocks)
    {
        const int nb = wave;
        v16h a = load_frag(qs + (nb * 16 + m16) * 40, 0, hlf);
#pragma unroll
        for (int mb = 0; mb < 4; ++mb) {
            v16h bf = load_frag(ks + (mb * 16 + m16) * 40, 0, hlf);
            v8f s = (v8f){0, 0, 0, 0, 0, 0, 0, 0};
            s = wmma_f16(a, bf, s);
#pragma unroll
            for (int v = 0; v < 8; ++v) {
                int nn = nb * 16 + v + 8 * hlf;
                int mm = mb * 16 + m16;
                sb[nn * 66 + mm] = s[v] * sc + bt[nn * 64 + mm];
            }
        }
    }
    __syncthreads();

    // row softmax in fp32, probs to f16 LDS
    if (tid < 64) {
        float mx = -1e30f;
        for (int j = 0; j < 64; ++j) mx = fmaxf(mx, sb[tid * 66 + j]);
        float sum = 0.0f;
        for (int j = 0; j < 64; ++j) {
            float e = __expf(sb[tid * 66 + j] - mx);
            sum += e;
            af[tid * 72 + j] = (_Float16)e;
        }
        float inv = 1.0f / sum;
        for (int j = 0; j < 64; ++j)
            af[tid * 72 + j] = (_Float16)((float)af[tid * 72 + j] * inv);
    }
    __syncthreads();

    // out = attn @ v   (64x64 @ 64x32, K=64 -> 2 WMMA steps)
    {
        const int nb = wave;
#pragma unroll
        for (int db = 0; db < 2; ++db) {
            v8f o = (v8f){0, 0, 0, 0, 0, 0, 0, 0};
#pragma unroll
            for (int kb = 0; kb < 2; ++kb) {
                v16h a  = load_frag(af + (nb * 16 + m16) * 72, kb * 32, hlf);
                v16h bf = load_frag(vsT + (db * 16 + m16) * 72, kb * 32, hlf);
                o = wmma_f16(a, bf, o);
            }
#pragma unroll
            for (int v = 0; v < 8; ++v) {
                int t = nb * 16 + v + 8 * hlf;
                attnout[((long)b * 64 + t) * 256 + h * 32 + db * 16 + m16] = (_Float16)o[v];
            }
        }
    }
}

// ---------------------------------------------------------------------------
// Kernel 4: output projection.  out[131072x256] = attnout @ proj_w^T + proj_b.
// grid = 2048, block = 256 (8 waves), 64x256 block tile, 32x64 wave tile.
// ---------------------------------------------------------------------------
__global__ void proj_kernel(const _Float16* __restrict__ attnout,
                            const float* __restrict__ proj_w,   // [256][256]
                            const float* __restrict__ proj_b,   // [256]
                            float* __restrict__ out) {
    extern __shared__ char smem[];
    _Float16* as = (_Float16*)smem;     // [64][40]
    _Float16* bs = as + 64 * 40;        // [256][40]
    const int tid = threadIdx.x, lane = tid & 31, wave = tid >> 5;
    const int m16 = lane & 15, hlf = lane >> 4;
    const int rp = wave & 1, cq = wave >> 1;
    const long rowBase = (long)blockIdx.x * 64;

    v8f acc[8];
#pragma unroll
    for (int t = 0; t < 8; ++t) acc[t] = (v8f){0, 0, 0, 0, 0, 0, 0, 0};

    for (int k0 = 0; k0 < 256; k0 += 32) {
        for (int c = tid; c < 256; c += 256) {       // A tile, already f16
            int r = c >> 2, cc = (c & 3) * 8;
            const _Float16* src = attnout + (rowBase + r) * 256 + k0 + cc;
            if (k0 + 32 < 256) __builtin_prefetch(src + 32, 0, 3);
            *(v8h*)(as + r * 40 + cc) = *(const v8h*)src;
        }
        for (int c = tid; c < 2048; c += 256) {      // B tile: proj_w rows, f32->f16
            int r = c >> 3, cc = (c & 7) * 4;
            const float* src = proj_w + (long)r * 256 + k0 + cc;
            float4 f = *(const float4*)src;
            if (k0 + 32 < 256) __builtin_prefetch(src + 32, 0, 3);
            _Float16* d = bs + r * 40 + cc;
            d[0] = (_Float16)f.x; d[1] = (_Float16)f.y;
            d[2] = (_Float16)f.z; d[3] = (_Float16)f.w;
        }
        __syncthreads();
        v16h a0 = load_frag(as + (rp * 32 + m16) * 40, 0, hlf);
        v16h a1 = load_frag(as + (rp * 32 + 16 + m16) * 40, 0, hlf);
#pragma unroll
        for (int t = 0; t < 4; ++t) {
            v16h b = load_frag(bs + (cq * 64 + t * 16 + m16) * 40, 0, hlf);
            acc[t]     = wmma_f16(a0, b, acc[t]);
            acc[4 + t] = wmma_f16(a1, b, acc[4 + t]);
        }
        __syncthreads();
    }

#pragma unroll
    for (int j = 0; j < 8; ++j) {
        int col = cq * 64 + (j & 3) * 16 + m16;
        float pb = proj_b[col];
#pragma unroll
        for (int v = 0; v < 8; ++v) {
            int r = rp * 32 + (j >> 2) * 16 + v + 8 * hlf;
            out[(rowBase + r) * 256 + col] = acc[j][v] + pb;
        }
    }
}

// ---------------------------------------------------------------------------
extern "C" void kernel_launch(void* const* d_in, const int* in_sizes, int n_in,
                              void* d_out, int out_size, void* d_ws, size_t ws_size,
                              hipStream_t stream) {
    (void)in_sizes; (void)n_in; (void)out_size; (void)ws_size;

    const float* x        = (const float*)d_in[0];
    const float* qkv_w    = (const float*)d_in[1];
    const float* q_bias   = (const float*)d_in[2];
    const float* v_bias   = (const float*)d_in[3];
    const float* scale    = (const float*)d_in[4];
    const float* cpb_w1   = (const float*)d_in[5];
    const float* cpb_b1   = (const float*)d_in[6];
    const float* cpb_w2   = (const float*)d_in[7];
    const float* proj_w   = (const float*)d_in[8];
    const float* proj_b   = (const float*)d_in[9];
    const float* rel_tbl  = (const float*)d_in[10];
    const int*   rel_idx  = (const int*)d_in[11];
    float* out = (float*)d_out;

    // workspace layout (bytes): biasTab 128KB | qbuf 64MB | kbuf 64MB | vbuf 64MB | attnout 64MB
    char* ws = (char*)d_ws;
    const size_t QKV_BYTES = (size_t)BATCH * HEADS * NTOK * HD * sizeof(_Float16); // 64 MiB
    float*    biasTab = (float*)ws;
    _Float16* qbuf    = (_Float16*)(ws + 131072);
    _Float16* kbuf    = (_Float16*)(ws + 131072 + QKV_BYTES);
    _Float16* vbuf    = (_Float16*)(ws + 131072 + 2 * QKV_BYTES);
    _Float16* attnout = (_Float16*)(ws + 131072 + 3 * QKV_BYTES);

    cpb_kernel<<<1, 256, 0, stream>>>(rel_tbl, cpb_w1, cpb_b1, cpb_w2, rel_idx, biasTab);

    qkv_kernel<<<dim3(BATCH, 3), 256, 33792, stream>>>(
        x, qkv_w, q_bias, v_bias, qbuf, kbuf, vbuf);

    attn_kernel<<<dim3(BATCH, HEADS), 128, 40960, stream>>>(
        qbuf, kbuf, vbuf, biasTab, scale, attnout);

    proj_kernel<<<BATCH, 256, 25600, stream>>>(attnout, proj_w, proj_b, out);
}
